// LuongAttention_71949292143257
// MI455X (gfx1250) — matrix operand order repaired and checked by reading, not compiled, not run on hardware
//
#include <hip/hip_runtime.h>
#include <cstdint>

// ============================================================================
// Luong attention, MI455X (gfx1250, wave32).
// ~137 GFLOP of GEMM vs ~470 MB HBM traffic (~20 us @ 23.3 TB/s): all three
// GEMMs run on v_wmma_f32_16x16x32_bf16 (f32 accumulate); softmax stays f32.
// GEMM: block tile 128x256, BK=32, 8 waves in 2(M)x4(N) grid, 64x64 per wave
// (16 WMMA : 16 ds_load_b128 per K-step). Double-buffered LDS, one barrier
// per K-step, async global->LDS staging (ASYNCcnt) overlapped with compute.
// ============================================================================

typedef __bf16 bf16_t;
typedef __attribute__((ext_vector_type(16))) __bf16 v16bf;
typedef __attribute__((ext_vector_type(8)))  __bf16 v8bf;
typedef __attribute__((ext_vector_type(4)))  __bf16 v4bf;
typedef __attribute__((ext_vector_type(8)))  float  v8f;
typedef __attribute__((ext_vector_type(4)))  float  v4f;

#if __has_builtin(__builtin_amdgcn_global_load_async_to_lds_b128)
#define USE_ASYNC_LDS 1
// Builtin signature (from hipcc diagnostic):
//   (int __vector(4) __device__*, int __vector(4) __shared__*, Imm off, Imm cpol)
typedef int v4i_t __attribute__((vector_size(4 * sizeof(int))));
typedef __attribute__((address_space(1))) v4i_t* gv4i_p;
typedef __attribute__((address_space(3))) v4i_t* lv4i_p;

__device__ __forceinline__ void async_copy_b128(const bf16_t* g, bf16_t* l) {
  __builtin_amdgcn_global_load_async_to_lds_b128(
      (gv4i_p)(v4i_t*)(void*)const_cast<bf16_t*>(g),
      (lv4i_p)(v4i_t*)(void*)l, 0, 0);
}

__device__ __forceinline__ void wait_asynccnt0() {
#if __has_builtin(__builtin_amdgcn_s_wait_asynccnt)
  __builtin_amdgcn_s_wait_asynccnt(0);
#else
  asm volatile("s_wait_asynccnt 0x0" ::: "memory");
#endif
}
#else
#define USE_ASYNC_LDS 0
#endif

union FragAB { v16bf v; v8bf h[2]; };
union PackBF2 { bf16_t h[2]; uint32_t u; };

// ---------------------------------------------------------------------------
// f32 -> bf16 elementwise convert (vectorized x4)
// ---------------------------------------------------------------------------
__global__ __launch_bounds__(256)
void k_f32_to_bf16(const float* __restrict__ in, bf16_t* __restrict__ out, long n4) {
  long i = (long)blockIdx.x * blockDim.x + threadIdx.x;
  long stride = (long)gridDim.x * blockDim.x;
  for (; i < n4; i += stride) {
    v4f f = ((const v4f*)in)[i];
    v4bf o;
#pragma unroll
    for (int j = 0; j < 4; ++j) o[j] = (bf16_t)f[j];
    ((v4bf*)out)[i] = o;
  }
}

// ---------------------------------------------------------------------------
// W [K][N] f32 -> WT [N][K] bf16 (tiled transpose via LDS)
// ---------------------------------------------------------------------------
__global__ __launch_bounds__(256)
void k_transpose_f32_to_bf16(const float* __restrict__ W, bf16_t* __restrict__ WT,
                             int K, int N) {
  __shared__ float tile[32][33];
  int n0 = blockIdx.x * 32, k0 = blockIdx.y * 32;
  int tx = threadIdx.x & 31, ty = threadIdx.x >> 5;  // 8 rows of 32
#pragma unroll
  for (int i = 0; i < 32; i += 8)
    tile[ty + i][tx] = W[(size_t)(k0 + ty + i) * N + (n0 + tx)];
  __syncthreads();
#pragma unroll
  for (int i = 0; i < 32; i += 8)
    WT[(size_t)(n0 + ty + i) * K + (k0 + tx)] = (bf16_t)tile[tx][ty + i];
}

// ---------------------------------------------------------------------------
// Generic bf16 WMMA GEMM: C[M][N] = A[M][K] * B[K][N] (+bias)
//   B_IS_KMAJOR == 0 : Bm is B^T, i.e. [N][K] row-major (K contiguous)
//   B_IS_KMAJOR == 1 : Bm is B,   i.e. [K][N] row-major (transposed into LDS)
// Block tile 128(M) x 256(N), BK=32, 256 threads = 8 waves (wave32),
// wave grid 2(M) x 4(N), each wave computes 64x64 = 4x4 WMMA tiles.
// Double-buffered LDS, one barrier per K-step. blockIdx.z = batch.
// ---------------------------------------------------------------------------
template <int B_IS_KMAJOR, int ADD_BIAS, int OUT_BF16>
__global__ __launch_bounds__(256)
void k_wmma_gemm(const bf16_t* __restrict__ A, const bf16_t* __restrict__ Bm,
                 const float* __restrict__ bias,
                 float* __restrict__ Cf, bf16_t* __restrict__ Cb,
                 int M, int N, int K, long strA, long strB, long strC) {
  A  += (size_t)blockIdx.z * strA;
  Bm += (size_t)blockIdx.z * strB;
  if (OUT_BF16) Cb += (size_t)blockIdx.z * strC;
  else          Cf += (size_t)blockIdx.z * strC;

  // row stride 40 halves (80 B): fragment reads tile all 64 banks conflict-free
  __shared__ bf16_t sA[2][128][40];
  __shared__ bf16_t sB[2][256][40];   // always stored as B^T: [n][k]

  const int m0 = blockIdx.y * 128;
  const int n0 = blockIdx.x * 256;

  const int tid   = threadIdx.x;
  const int lane  = tid & 31;
  const int wave  = tid >> 5;
  const int waveM = wave >> 2;     // 0..1
  const int waveN = wave & 3;      // 0..3
  const int hgrp  = lane >> 4;     // lane group 0/1
  const int l16   = lane & 15;

  const int rowA = tid >> 1;            // 0..127
  const int colA = (tid & 1) * 16;      // 0 or 16
  // mode-1 staging: lanes span K (conflict-free packed-dword LDS scatter)
  const int kkp = (tid & 15) * 2;       // even k within tile: 0..30
  const int nnp = (tid >> 4) * 8;       // 0..120 (second segment at +128)

  v8f acc[4][4];
#pragma unroll
  for (int i = 0; i < 4; ++i)
#pragma unroll
    for (int j = 0; j < 4; ++j)
#pragma unroll
      for (int r = 0; r < 8; ++r) acc[i][j][r] = 0.0f;

  v8bf rA0{}, rA1{}, rB0{}, rB1{}, rB2{}, rB3{};
  v8bf rT0{}, rT1{}, rT2{}, rT3{};

  auto stage_issue = [&](int k0, int buf) {
    // ---- A tile 128x32 ----
    const bf16_t* srcA = A + (size_t)(m0 + rowA) * K + (k0 + colA);
#if USE_ASYNC_LDS
    {
      bf16_t* dA = &sA[buf][rowA][colA];
      async_copy_b128(srcA,     dA);
      async_copy_b128(srcA + 8, dA + 8);
    }
#else
    rA0 = *(const v8bf*)srcA;
    rA1 = *(const v8bf*)(srcA + 8);
#endif
    // ---- B tile -> B^T [256 n][32 k] ----
    if (B_IS_KMAJOR) {
      // source [K][N]: each thread reads a kk-pair of two 8-wide n chunks
      const bf16_t* s0 = Bm + (size_t)(k0 + kkp) * N + (n0 + nnp);
      const bf16_t* s1 = Bm + (size_t)(k0 + kkp + 1) * N + (n0 + nnp);
      rT0 = *(const v8bf*)s0;
      rT1 = *(const v8bf*)s1;
      rT2 = *(const v8bf*)(s0 + 128);
      rT3 = *(const v8bf*)(s1 + 128);
    } else {
      const bf16_t* srcB0 = Bm + (size_t)(n0 + rowA) * K + (k0 + colA);
      const bf16_t* srcB1 = Bm + (size_t)(n0 + rowA + 128) * K + (k0 + colA);
#if USE_ASYNC_LDS
      bf16_t* dB0 = &sB[buf][rowA][colA];
      bf16_t* dB1 = &sB[buf][rowA + 128][colA];
      async_copy_b128(srcB0,     dB0);
      async_copy_b128(srcB0 + 8, dB0 + 8);
      async_copy_b128(srcB1,     dB1);
      async_copy_b128(srcB1 + 8, dB1 + 8);
#else
      rB0 = *(const v8bf*)srcB0;
      rB1 = *(const v8bf*)(srcB0 + 8);
      rB2 = *(const v8bf*)srcB1;
      rB3 = *(const v8bf*)(srcB1 + 8);
#endif
    }
  };

  auto stage_commit = [&](int buf) {
#if !USE_ASYNC_LDS
    *(v8bf*)&sA[buf][rowA][colA]     = rA0;
    *(v8bf*)&sA[buf][rowA][colA + 8] = rA1;
#endif
    if (B_IS_KMAJOR) {
      // pack {k, k+1} into one dword: lanes 0-15 hit 16 consecutive banks,
      // lanes 16-31 a disjoint set -> conflict-free transpose scatter
#pragma unroll
      for (int j = 0; j < 8; ++j) {
        PackBF2 p0; p0.h[0] = rT0[j]; p0.h[1] = rT1[j];
        *(uint32_t*)&sB[buf][nnp + j][kkp] = p0.u;
        PackBF2 p1; p1.h[0] = rT2[j]; p1.h[1] = rT3[j];
        *(uint32_t*)&sB[buf][nnp + 128 + j][kkp] = p1.u;
      }
    } else {
#if !USE_ASYNC_LDS
      *(v8bf*)&sB[buf][rowA][colA]           = rB0;
      *(v8bf*)&sB[buf][rowA][colA + 8]       = rB1;
      *(v8bf*)&sB[buf][rowA + 128][colA]     = rB2;
      *(v8bf*)&sB[buf][rowA + 128][colA + 8] = rB3;
#endif
    }
  };

  auto wait_stage = [&]() {
#if USE_ASYNC_LDS
    wait_asynccnt0();
#endif
    __syncthreads();
  };

  auto compute = [&](int buf) {
    // A 16-bit layout: lanes 0-15 hold K {0..7,16..23}, lanes 16-31 K {8..15,24..31}
    FragAB fa[4];
#pragma unroll
    for (int i = 0; i < 4; ++i) {
      int mr = waveM * 64 + i * 16 + l16;
      fa[i].h[0] = *(const v8bf*)&sA[buf][mr][hgrp * 8];
      fa[i].h[1] = *(const v8bf*)&sA[buf][mr][16 + hgrp * 8];
    }
    // B 16-bit layout: same as A, lanes 0-15 hold K {0..7,16..23}, lanes 16-31 K {8..15,24..31}
    FragAB fb[4];
#pragma unroll
    for (int j = 0; j < 4; ++j) {
      int nc = waveN * 64 + j * 16 + l16;
      fb[j].h[0] = *(const v8bf*)&sB[buf][nc][hgrp * 8];
      fb[j].h[1] = *(const v8bf*)&sB[buf][nc][16 + hgrp * 8];
    }
#pragma unroll
    for (int i = 0; i < 4; ++i)
#pragma unroll
      for (int j = 0; j < 4; ++j)
        acc[i][j] = __builtin_amdgcn_wmma_f32_16x16x32_bf16(
            false, fa[i].v, false, fb[j].v, (short)0, acc[i][j], false, false);
  };

  // -------- software pipeline: one barrier per K-step --------
  stage_issue(0, 0);
  stage_commit(0);
  wait_stage();

  int buf = 0;
  for (int k0 = 0; k0 < K; k0 += 32) {
    const bool more = (k0 + 32) < K;
    if (more) {
      stage_issue(k0 + 32, buf ^ 1);
      if (k0 + 64 < K)  // stay in-bounds: WGP-scope prefetch is non-speculative
        __builtin_prefetch(A + (size_t)(m0 + rowA) * K + (k0 + 64 + colA), 0, 3);
    }
    compute(buf);
    if (more) {
      stage_commit(buf ^ 1);
      wait_stage();
    }
    buf ^= 1;
  }

  // ---- epilogue: acc VGPR r / lane l -> (M = 8*(l>>4)+r, N = l&15) ----
#pragma unroll
  for (int i = 0; i < 4; ++i) {
#pragma unroll
    for (int j = 0; j < 4; ++j) {
      int gn = n0 + waveN * 64 + j * 16 + l16;
      float bv = ADD_BIAS ? bias[gn] : 0.0f;
#pragma unroll
      for (int r = 0; r < 8; ++r) {
        int gm = m0 + waveM * 64 + i * 16 + hgrp * 8 + r;
        float v = acc[i][j][r] + bv;
        if (OUT_BF16) Cb[(size_t)gm * N + gn] = (bf16_t)v;
        else          Cf[(size_t)gm * N + gn] = v;
      }
    }
  }
}

// ---------------------------------------------------------------------------
// Row softmax over Tk=2048; one block per (b,q) row. Writes normalized f32
// (d_out alignment) in place and a bf16 copy for the context GEMM.
// ---------------------------------------------------------------------------
__global__ __launch_bounds__(256)
void k_softmax(float* __restrict__ sc, bf16_t* __restrict__ ab, int Tk) {
  long row = blockIdx.x;
  float*  p = sc + row * (long)Tk;
  bf16_t* q = ab + row * (long)Tk;
  __shared__ float red[256];
  int t = threadIdx.x;
  float vals[8];
  float m = -3.402823e38f;
#pragma unroll
  for (int i = 0; i < 8; ++i) { vals[i] = p[t + i * 256]; m = fmaxf(m, vals[i]); }
  red[t] = m; __syncthreads();
  for (int s = 128; s > 0; s >>= 1) {
    if (t < s) red[t] = fmaxf(red[t], red[t + s]);
    __syncthreads();
  }
  m = red[0]; __syncthreads();
  float sum = 0.f;
#pragma unroll
  for (int i = 0; i < 8; ++i) { vals[i] = __expf(vals[i] - m); sum += vals[i]; }
  red[t] = sum; __syncthreads();
  for (int s = 128; s > 0; s >>= 1) {
    if (t < s) red[t] += red[t + s];
    __syncthreads();
  }
  float inv = 1.0f / red[0];
#pragma unroll
  for (int i = 0; i < 8; ++i) {
    float r = vals[i] * inv;
    p[t + i * 256] = r;
    q[t + i * 256] = (bf16_t)r;
  }
}

// ---------------------------------------------------------------------------
// launch
// ---------------------------------------------------------------------------
extern "C" void kernel_launch(void* const* d_in, const int* in_sizes, int n_in,
                              void* d_out, int out_size, void* d_ws, size_t ws_size,
                              hipStream_t stream) {
  const int B = 16, Tq = 512, Tk = 2048, D = 1024;

  const float* dec  = (const float*)d_in[0];  // [B][Tq][D]
  const float* enc  = (const float*)d_in[1];  // [B][Tk][D]
  const float* Wk   = (const float*)d_in[2];  // [D][D]
  const float* bias = (const float*)d_in[3];  // [D]
  float* out = (float*)d_out;                 // ctx [B*Tq*D] then align [B*Tq*Tk]

  // workspace layout (~187 MB, all 32B-aligned)
  char* ws = (char*)d_ws;
  bf16_t* dec_b = (bf16_t*)ws; ws += (size_t)B * Tq * D * sizeof(bf16_t);
  bf16_t* enc_b = (bf16_t*)ws; ws += (size_t)B * Tk * D * sizeof(bf16_t);
  bf16_t* wT    = (bf16_t*)ws; ws += (size_t)D * D * sizeof(bf16_t);
  bf16_t* proj  = (bf16_t*)ws; ws += (size_t)B * Tk * D * sizeof(bf16_t);
  bf16_t* alnb  = (bf16_t*)ws;

  float* ctx = out;
  float* aln = out + (size_t)B * Tq * D;

  // 1) precision staging
  {
    long n4d = (long)B * Tq * D / 4;
    long n4e = (long)B * Tk * D / 4;
    k_f32_to_bf16<<<(int)((n4d + 255) / 256), 256, 0, stream>>>(dec, dec_b, n4d);
    k_f32_to_bf16<<<(int)((n4e + 255) / 256), 256, 0, stream>>>(enc, enc_b, n4e);
    k_transpose_f32_to_bf16<<<dim3(D / 32, D / 32), 256, 0, stream>>>(Wk, wT, D, D);
  }

  // 2) proj = enc @ W + bias  (one big GEMM: M = B*Tk, W shared across batch)
  k_wmma_gemm<0, 1, 1><<<dim3(D / 256, (B * Tk) / 128, 1), 256, 0, stream>>>(
      enc_b, wT, bias, nullptr, proj, B * Tk, D, D, 0, 0, 0);

  // 3) score = dec @ proj^T  (proj rows are already K-contiguous B^T form)
  k_wmma_gemm<0, 0, 0><<<dim3(Tk / 256, Tq / 128, B), 256, 0, stream>>>(
      dec_b, proj, nullptr, aln, nullptr, Tq, Tk, D,
      (long)Tq * D, (long)Tk * D, (long)Tq * Tk);

  // 4) softmax over Tk, in place on d_out; bf16 copy for the context GEMM
  k_softmax<<<B * Tq, 256, 0, stream>>>(aln, alnb, Tk);

  // 5) ctx = align @ enc  (enc is [K][N] -> transpose while staging into LDS)
  k_wmma_gemm<1, 0, 0><<<dim3(D / 256, Tq / 128, B), 256, 0, stream>>>(
      alnb, enc_b, nullptr, ctx, nullptr, Tq, D, Tk,
      (long)Tq * Tk, (long)Tk * D, (long)Tq * D);
}
